// Interaction_31190052503575
// MI455X (gfx1250) — matrix operand order repaired and checked by reading, not compiled, not run on hardware
//
#include <hip/hip_runtime.h>
#include <hip/hip_fp16.h>

typedef __attribute__((ext_vector_type(16))) _Float16 v16h;
typedef __attribute__((ext_vector_type(4)))  _Float16 v4h;
typedef __attribute__((ext_vector_type(8)))  float    v8f;

#define HD   128   // hidden channels
#define NBIL 8     // num bilinear
#define NRAD 6     // num radial
#define NSR  42    // NS*NR

__device__ __forceinline__ float silu_f(float v) {
    return v / (1.0f + __expf(-v));
}

union F16Frag { v16h v; uint4 q[2]; };

// ---------------------------------------------------------------------------
// Kernel 0: one-time convert W_bil (fp32 [n,j,k]) -> f16 slabs [j][n][k]
// so the bilinear kernel stages each 32KB B_j slab with async b128 copies.
// ---------------------------------------------------------------------------
__global__ __launch_bounds__(256) void wbil_cvt_kernel(
    const float* __restrict__ W_bil, _Float16* __restrict__ wbil_h)
{
    int id = (blockIdx.x * 256 + threadIdx.x) * 4;   // 4 consecutive k
    if (id >= NBIL * HD * HD) return;
    int j = id >> 14;
    int rem = id & 16383;
    int n = rem >> 7, k = rem & 127;
    float4 w = *(const float4*)(W_bil + (size_t)n * (NBIL * HD) + j * HD + k);
    v4h wh = { (_Float16)w.x, (_Float16)w.y, (_Float16)w.z, (_Float16)w.w };
    *(v4h*)(wbil_h + id) = wh;
}

// ---------------------------------------------------------------------------
// Kernel 1: per-edge MLPs.
//   xkj_h[e,:] = f16( silu(x[e] @ Wkj^T + bkj) * (rbf[e] @ Wrbf^T) )
//   xji_f[e,:] =      silu(x[e] @ Wji^T + bji)
// ---------------------------------------------------------------------------
__global__ __launch_bounds__(256) void edge_mlp_kernel(
    const float* __restrict__ x, const float* __restrict__ rbf,
    const float* __restrict__ W_rbf,
    const float* __restrict__ W_kj, const float* __restrict__ b_kj,
    const float* __restrict__ W_ji, const float* __restrict__ b_ji,
    _Float16* __restrict__ xkj_h, float* __restrict__ xji_f, int E)
{
    __shared__ _Float16 s_w[HD * HD];     // 32 KB, reused (kj then ji)
    __shared__ float    s_wrbf[HD * NRAD];
    __shared__ float    s_bkj[HD], s_bji[HD];
    __shared__ float    s_rbf[8][16 * NRAD];

    const int tid  = threadIdx.x;
    const int wave = tid >> 5, lane = tid & 31;
    const int tile = blockIdx.x * 8 + wave;
    const bool valid = tile < (E >> 4);
    const int e0 = tile * 16;
    const int m = lane & 15, g = lane >> 4;

    for (int i = tid; i < HD * NRAD; i += 256) s_wrbf[i] = W_rbf[i];
    if (tid < HD) { s_bkj[tid] = b_kj[tid]; s_bji[tid] = b_ji[tid]; }
    if (valid && lane < 16) {
        #pragma unroll
        for (int r = 0; r < NRAD; ++r)
            s_rbf[wave][lane * NRAD + r] = rbf[(e0 + lane) * NRAD + r];
    }

    // ---- stage Wkj (f16) ----
    for (int i = tid * 4; i < HD * HD; i += 1024) {
        float4 w = *(const float4*)(W_kj + i);
        v4h wh = { (_Float16)w.x, (_Float16)w.y, (_Float16)w.z, (_Float16)w.w };
        *(v4h*)(s_w + i) = wh;
    }
    __syncthreads();

    // A fragments from global x (fp32 -> f16), reused for both GEMMs.
    v16h af[4];
    if (valid) {
        #pragma unroll
        for (int kt = 0; kt < 4; ++kt) {
            const float* px = x + (size_t)(e0 + m) * HD + kt * 32;
            union { float4 f[4]; float s[16]; } u;
            u.f[0] = *(const float4*)(px + 8 * g);
            u.f[1] = *(const float4*)(px + 8 * g + 4);
            u.f[2] = *(const float4*)(px + 16 + 8 * g);
            u.f[3] = *(const float4*)(px + 16 + 8 * g + 4);
            #pragma unroll
            for (int i = 0; i < 16; ++i) af[kt][i] = (_Float16)u.s[i];
        }
        v8f acc[8] = {};
        #pragma unroll
        for (int kt = 0; kt < 4; ++kt) {
            #pragma unroll
            for (int nt = 0; nt < 8; ++nt) {
                F16Frag bf;
                const _Float16* bp = s_w + (nt * 16 + m) * HD + kt * 32 + 8 * g;
                bf.q[0] = *(const uint4*)(bp);
                bf.q[1] = *(const uint4*)(bp + 16);
                acc[nt] = __builtin_amdgcn_wmma_f32_16x16x32_f16(
                    false, af[kt], false, bf.v, (short)0, acc[nt], false, false);
            }
        }
        #pragma unroll
        for (int nt = 0; nt < 8; ++nt) {
            int n = nt * 16 + m;
            #pragma unroll
            for (int v = 0; v < 8; ++v) {
                int row = 8 * g + v;
                float rh = 0.0f;
                #pragma unroll
                for (int r = 0; r < NRAD; ++r)
                    rh += s_rbf[wave][row * NRAD + r] * s_wrbf[n * NRAD + r];
                float val = silu_f(acc[nt][v] + s_bkj[n]) * rh;
                xkj_h[(size_t)(e0 + row) * HD + n] = (_Float16)val;
            }
        }
    }
    __syncthreads();

    // ---- stage Wji (f16) ----
    for (int i = tid * 4; i < HD * HD; i += 1024) {
        float4 w = *(const float4*)(W_ji + i);
        v4h wh = { (_Float16)w.x, (_Float16)w.y, (_Float16)w.z, (_Float16)w.w };
        *(v4h*)(s_w + i) = wh;
    }
    __syncthreads();

    if (valid) {
        v8f acc[8] = {};
        #pragma unroll
        for (int kt = 0; kt < 4; ++kt) {
            #pragma unroll
            for (int nt = 0; nt < 8; ++nt) {
                F16Frag bf;
                const _Float16* bp = s_w + (nt * 16 + m) * HD + kt * 32 + 8 * g;
                bf.q[0] = *(const uint4*)(bp);
                bf.q[1] = *(const uint4*)(bp + 16);
                acc[nt] = __builtin_amdgcn_wmma_f32_16x16x32_f16(
                    false, af[kt], false, bf.v, (short)0, acc[nt], false, false);
            }
        }
        #pragma unroll
        for (int nt = 0; nt < 8; ++nt) {
            int n = nt * 16 + m;
            #pragma unroll
            for (int v = 0; v < 8; ++v) {
                int row = 8 * g + v;
                xji_f[(size_t)(e0 + row) * HD + n] = silu_f(acc[nt][v] + s_bji[n]);
            }
        }
    }
}

// ---------------------------------------------------------------------------
// Kernel 2: sbf_b[t,j] = sbf[t,:] @ W_sbf[j,:]   (bandwidth bound, 252 MB in)
// ---------------------------------------------------------------------------
__global__ __launch_bounds__(256) void sbf_proj_kernel(
    const float* __restrict__ sbf, const float* __restrict__ W_sbf,
    float* __restrict__ sbfb, int T)
{
    __shared__ float s_w[NBIL * NSR];
    for (int i = threadIdx.x; i < NBIL * NSR; i += 256) s_w[i] = W_sbf[i];
    __syncthreads();
    int t = blockIdx.x * 256 + threadIdx.x;
    if (t >= T) return;
    float row[NSR];
    #pragma unroll
    for (int i = 0; i < NSR; ++i) row[i] = sbf[(size_t)t * NSR + i];
    #pragma unroll
    for (int j = 0; j < NBIL; ++j) {
        float a = 0.0f;
        #pragma unroll
        for (int i = 0; i < NSR; ++i) a += row[i] * s_w[j * NSR + i];
        sbfb[(size_t)t * NBIL + j] = a;
    }
}

// ---------------------------------------------------------------------------
// Async stage of one 32KB f16 B_j slab into LDS:
// 256 threads x 8 chunks x 16B = 32768 B, via GLOBAL_LOAD_ASYNC_TO_LDS_B128
// (tracked by ASYNCcnt; completed with s_wait_asynccnt + barrier).
// ---------------------------------------------------------------------------
__device__ __forceinline__ void stage_bj_async(
    const _Float16* __restrict__ src, _Float16* dst_lds, int tid)
{
    uint32_t lds_base = (uint32_t)(size_t)(void*)dst_lds;
    #pragma unroll
    for (int c = 0; c < 8; ++c) {
        uint32_t off = (uint32_t)(tid + c * 256) * 8;      // halves
        uint64_t ga  = (uint64_t)(size_t)(src + off);
        uint32_t la  = lds_base + off * 2;                 // bytes
        asm volatile("global_load_async_to_lds_b128 %0, %1, off"
                     :: "v"(la), "v"(ga) : "memory");
    }
}

__device__ __forceinline__ void wait_async_all() {
    asm volatile("s_wait_asynccnt 0" ::: "memory");
}

// ---------------------------------------------------------------------------
// Kernel 3: bilinear + gather + segment-sum (the 393 GFLOP hot spot).
// Per wave: one 16-triplet tile. A fragments (gathered x_kj rows, f16) live
// in 32 VGPRs, loaded once from L2-resident global and rescaled per j with
// v_pk_mul_f16. B_j slabs double-buffered in LDS via async loads, overlapping
// the next j's 32KB stage with the current j's 32 WMMAs.
// Scatter via global_atomic_add_f32 into L2-resident agg[E,128].
// ---------------------------------------------------------------------------
__global__ __launch_bounds__(256) void bilinear_kernel(
    const _Float16* __restrict__ xkj_h, const float* __restrict__ sbfb,
    const int* __restrict__ idx_kj, const int* __restrict__ idx_ji,
    const _Float16* __restrict__ wbil_h, float* __restrict__ agg, int T)
{
    __shared__ _Float16 s_b[2][HD * HD];    // 2 x 32 KB double buffer
    const int tid = threadIdx.x, wave = tid >> 5, lane = tid & 31;
    const int tile = blockIdx.x * 8 + wave;
    const bool valid = tile < (T >> 4);
    const int t0 = tile * 16;
    const int m = lane & 15, g = lane >> 4;

    // kick off j=0 B slab while we fetch A / scales / indices
    stage_bj_async(wbil_h, s_b[0], tid);

    v16h af[4];
    float sreg[8];
    int   earr[8];
    if (valid) {
        // A fragments straight from gathered global row (f16, L2 resident):
        // lane covers row m with K pattern {kt*32+8g .. +7} and {+16}.
        int e = idx_kj[t0 + m];
        const _Float16* prow = xkj_h + (size_t)e * HD;
        #pragma unroll
        for (int kt = 0; kt < 4; ++kt) {
            F16Frag a;
            a.q[0] = *(const uint4*)(prow + kt * 32 + 8 * g);
            a.q[1] = *(const uint4*)(prow + kt * 32 + 16 + 8 * g);
            af[kt] = a.v;
        }
        float4 s0 = *(const float4*)(sbfb + (size_t)(t0 + m) * NBIL);
        float4 s1 = *(const float4*)(sbfb + (size_t)(t0 + m) * NBIL + 4);
        sreg[0] = s0.x; sreg[1] = s0.y; sreg[2] = s0.z; sreg[3] = s0.w;
        sreg[4] = s1.x; sreg[5] = s1.y; sreg[6] = s1.z; sreg[7] = s1.w;
        int4 i0 = *(const int4*)(idx_ji + t0 + 8 * g);
        int4 i1 = *(const int4*)(idx_ji + t0 + 8 * g + 4);
        earr[0] = i0.x; earr[1] = i0.y; earr[2] = i0.z; earr[3] = i0.w;
        earr[4] = i1.x; earr[5] = i1.y; earr[6] = i1.z; earr[7] = i1.w;
    }

    wait_async_all();
    __syncthreads();

    v8f acc[8] = {};

    for (int j = 0; j < NBIL; ++j) {
        const int cur = j & 1;
        if (j < NBIL - 1)   // overlap next slab stage with this j's WMMAs
            stage_bj_async(wbil_h + (size_t)(j + 1) * (HD * HD), s_b[1 - cur], tid);

        if (valid) {
            _Float16 sj = (_Float16)sreg[j];
            #pragma unroll
            for (int kt = 0; kt < 4; ++kt) {
                v16h av = af[kt] * sj;     // fold sbf_b scale (v_pk_mul_f16)
                #pragma unroll
                for (int nt = 0; nt < 8; ++nt) {
                    F16Frag b;
                    const _Float16* bp = s_b[cur] + (nt * 16 + m) * HD + kt * 32 + 8 * g;
                    b.q[0] = *(const uint4*)(bp);
                    b.q[1] = *(const uint4*)(bp + 16);
                    acc[nt] = __builtin_amdgcn_wmma_f32_16x16x32_f16(
                        false, av, false, b.v, (short)0, acc[nt], false, false);
                }
            }
        }

        wait_async_all();
        __syncthreads();
    }

    if (valid) {
        #pragma unroll
        for (int nt = 0; nt < 8; ++nt) {
            int n = nt * 16 + m;
            #pragma unroll
            for (int v = 0; v < 8; ++v) {
                unsafeAtomicAdd(agg + (size_t)earr[v] * HD + n, acc[nt][v]);
            }
        }
    }
}

// ---------------------------------------------------------------------------
// Kernel 4: h = silu((x_ji + agg) @ W_lin^T + b_lin)
// ---------------------------------------------------------------------------
__global__ __launch_bounds__(256) void final_mlp_kernel(
    const float* __restrict__ xji_f, const float* __restrict__ agg,
    const float* __restrict__ W_lin, const float* __restrict__ b_lin,
    float* __restrict__ out, int E)
{
    __shared__ _Float16 s_w[HD * HD];
    __shared__ float    s_b[HD];
    const int tid  = threadIdx.x;
    const int wave = tid >> 5, lane = tid & 31;
    const int tile = blockIdx.x * 8 + wave;
    const bool valid = tile < (E >> 4);
    const int e0 = tile * 16;
    const int m = lane & 15, g = lane >> 4;

    for (int i = tid * 4; i < HD * HD; i += 1024) {
        float4 w = *(const float4*)(W_lin + i);
        v4h wh = { (_Float16)w.x, (_Float16)w.y, (_Float16)w.z, (_Float16)w.w };
        *(v4h*)(s_w + i) = wh;
    }
    if (tid < HD) s_b[tid] = b_lin[tid];
    __syncthreads();
    if (!valid) return;

    v16h af[4];
    #pragma unroll
    for (int kt = 0; kt < 4; ++kt) {
        const float* p1 = xji_f + (size_t)(e0 + m) * HD + kt * 32;
        const float* p2 = agg   + (size_t)(e0 + m) * HD + kt * 32;
        union { float4 f[4]; float s[16]; } u, w;
        u.f[0] = *(const float4*)(p1 + 8 * g);      w.f[0] = *(const float4*)(p2 + 8 * g);
        u.f[1] = *(const float4*)(p1 + 8 * g + 4);  w.f[1] = *(const float4*)(p2 + 8 * g + 4);
        u.f[2] = *(const float4*)(p1 + 16 + 8 * g); w.f[2] = *(const float4*)(p2 + 16 + 8 * g);
        u.f[3] = *(const float4*)(p1 + 16 + 8 * g + 4);
        w.f[3] = *(const float4*)(p2 + 16 + 8 * g + 4);
        #pragma unroll
        for (int i = 0; i < 16; ++i) af[kt][i] = (_Float16)(u.s[i] + w.s[i]);
    }

    v8f acc[8] = {};
    #pragma unroll
    for (int kt = 0; kt < 4; ++kt) {
        #pragma unroll
        for (int nt = 0; nt < 8; ++nt) {
            F16Frag bf;
            const _Float16* bp = s_w + (nt * 16 + m) * HD + kt * 32 + 8 * g;
            bf.q[0] = *(const uint4*)(bp);
            bf.q[1] = *(const uint4*)(bp + 16);
            acc[nt] = __builtin_amdgcn_wmma_f32_16x16x32_f16(
                false, af[kt], false, bf.v, (short)0, acc[nt], false, false);
        }
    }
    #pragma unroll
    for (int nt = 0; nt < 8; ++nt) {
        int n = nt * 16 + m;
        #pragma unroll
        for (int v = 0; v < 8; ++v) {
            int row = 8 * g + v;
            out[(size_t)(e0 + row) * HD + n] = silu_f(acc[nt][v] + s_b[n]);
        }
    }
}

// ---------------------------------------------------------------------------
extern "C" void kernel_launch(void* const* d_in, const int* in_sizes, int n_in,
                              void* d_out, int out_size, void* d_ws, size_t ws_size,
                              hipStream_t stream)
{
    const float* x      = (const float*)d_in[0];
    const float* rbf    = (const float*)d_in[1];
    const float* sbf    = (const float*)d_in[2];
    const int*   idx_kj = (const int*)d_in[3];
    const int*   idx_ji = (const int*)d_in[4];
    const float* W_rbf  = (const float*)d_in[5];
    const float* W_sbf  = (const float*)d_in[6];
    const float* W_kj   = (const float*)d_in[7];
    const float* b_kj   = (const float*)d_in[8];
    const float* W_ji   = (const float*)d_in[9];
    const float* b_ji   = (const float*)d_in[10];
    const float* W_bil  = (const float*)d_in[11];
    const float* W_lin  = (const float*)d_in[12];
    const float* b_lin  = (const float*)d_in[13];

    const int E = in_sizes[0] / HD;
    const int T = in_sizes[3];

    char* ws = (char*)d_ws;
    size_t o = 0;
    _Float16* wbil_h = (_Float16*)(ws + o); o += (size_t)NBIL * HD * HD * sizeof(_Float16);
    o = (o + 255) & ~(size_t)255;
    _Float16* xkj_h = (_Float16*)(ws + o); o += (size_t)E * HD * sizeof(_Float16);
    o = (o + 255) & ~(size_t)255;
    float* xji_f = (float*)(ws + o); o += (size_t)E * HD * sizeof(float);
    o = (o + 255) & ~(size_t)255;
    float* sbfb  = (float*)(ws + o); o += (size_t)T * NBIL * sizeof(float);
    o = (o + 255) & ~(size_t)255;
    float* agg   = (float*)(ws + o); o += (size_t)E * HD * sizeof(float);

    const int etiles = E / 16;
    const int ttiles = T / 16;
    const int eblocks = (etiles + 7) / 8;
    const int tblocks = (ttiles + 7) / 8;

    wbil_cvt_kernel<<<(NBIL * HD * HD / 4 + 255) / 256, 256, 0, stream>>>(W_bil, wbil_h);
    edge_mlp_kernel<<<eblocks, 256, 0, stream>>>(
        x, rbf, W_rbf, W_kj, b_kj, W_ji, b_ji, xkj_h, xji_f, E);
    sbf_proj_kernel<<<(T + 255) / 256, 256, 0, stream>>>(sbf, W_sbf, sbfb, T);
    hipMemsetAsync(agg, 0, (size_t)E * HD * sizeof(float), stream);
    bilinear_kernel<<<tblocks, 256, 0, stream>>>(
        xkj_h, sbfb, idx_kj, idx_ji, wbil_h, agg, T);
    final_mlp_kernel<<<eblocks, 256, 0, stream>>>(
        xji_f, agg, W_lin, b_lin, (float*)d_out, E);
}
